// MHSelfAttn_38233798869259
// MI455X (gfx1250) — matrix operand order repaired and checked by reading, not compiled, not run on hardware
//
#include <hip/hip_runtime.h>
#include <hip/hip_bf16.h>
#include <math.h>

typedef __bf16 bf16;
typedef __attribute__((ext_vector_type(16))) __bf16 v16bf;
typedef __attribute__((ext_vector_type(8)))  __bf16 v8bf;
typedef __attribute__((ext_vector_type(8)))  float  v8f;

#define DIM    256
#define NTOK   1024
#define NHEAD  4
#define DH     64
#define NBATCH 16
#define SROW   36   // padded fp32 score row stride (bank-conflict avoidance)
#define PROW   40   // padded bf16 P row stride (multiple of 8 for 16B-aligned loads)

__device__ __forceinline__ bf16 f2bf(float f) {
    union { float f; unsigned u; } c; c.f = f;
    unsigned r = c.u + 0x7FFFu + ((c.u >> 16) & 1u);   // round-to-nearest-even
    union { unsigned short s; bf16 b; } o; o.s = (unsigned short)(r >> 16);
    return o.b;
}

// ---------------------------------------------------------------------------
// Kernel 1: GroupNorm over (8 channels x 1024 spatial) per (batch, group);
// writes bf16 activations token-major: xnb[b][t][c]
// ---------------------------------------------------------------------------
__global__ void gn_kernel(const float* __restrict__ x, const float* __restrict__ scale,
                          const float* __restrict__ bias, bf16* __restrict__ xnb) {
    __shared__ float s_sum[256], s_sq[256];
    int b = blockIdx.x >> 5;
    int g = blockIdx.x & 31;
    const float* xp = x + (size_t)b * DIM * NTOK + (size_t)g * 8 * NTOK;
    float sum = 0.f, sq = 0.f;
    for (int i = threadIdx.x; i < 8192; i += 256) { float v = xp[i]; sum += v; sq += v * v; }
    s_sum[threadIdx.x] = sum; s_sq[threadIdx.x] = sq;
    __syncthreads();
    for (int s = 128; s > 0; s >>= 1) {
        if (threadIdx.x < s) { s_sum[threadIdx.x] += s_sum[threadIdx.x + s];
                               s_sq[threadIdx.x]  += s_sq[threadIdx.x + s]; }
        __syncthreads();
    }
    float mean = s_sum[0] * (1.0f / 8192.0f);
    float var  = s_sq[0]  * (1.0f / 8192.0f) - mean * mean;
    float rstd = rsqrtf(var + 1e-6f);
    for (int i = threadIdx.x; i < 8192; i += 256) {
        int cl = i >> 10, t = i & 1023;
        int c = g * 8 + cl;
        float v = (xp[i] - mean) * rstd * scale[c] + bias[c];
        xnb[((size_t)b * NTOK + t) * DIM + c] = f2bf(v);
    }
}

// ---------------------------------------------------------------------------
// Kernel 2: pack the 4 weights (256x256 fp32) into WMMA B-fragment lane order.
// B frag (32x16 K x N): lane L holds k = (L/16)*16 + e, n = L%16  (e = 0..15)
// frag[((w*8+kt)*16+nt)*32 + lane][e].  Wq gets the 1/sqrt(C)=1/16 scale folded in.
// ---------------------------------------------------------------------------
__global__ void pack_kernel(const float* __restrict__ Wq, const float* __restrict__ Wk,
                            const float* __restrict__ Wv, const float* __restrict__ Wp,
                            bf16* __restrict__ frag) {
    int tid  = blockIdx.x * 256 + threadIdx.x;   // 16384 threads
    int lane = tid & 31;
    int nt   = (tid >> 5)  & 15;
    int kt   = (tid >> 9)  & 7;
    int w    = (tid >> 12) & 3;
    const float* W = (w == 0) ? Wq : (w == 1) ? Wk : (w == 2) ? Wv : Wp;
    float sc = (w == 0) ? 0.0625f : 1.0f;
    int n = nt * 16 + (lane & 15);
    int kbase = kt * 32 + (lane >> 4) * 16;
    bf16* dst = frag + ((size_t)((w * 8 + kt) * 16 + nt) * 32 + lane) * 16;
    for (int e = 0; e < 16; e++)
        dst[e] = f2bf(W[(size_t)(kbase + e) * DIM + n] * sc);
}

// ---------------------------------------------------------------------------
// Kernel 3: QKV projection GEMM (per batch: 1024x256 @ 256x256, x3).
// Q,K written token-major per (b,head): [tok][d]; V written d-major: [d][tok].
// The three projections are fully specialized: no branches in the hot loops.
// ---------------------------------------------------------------------------
__device__ __forceinline__ v8f gemm_tile(const v16bf a[8], const bf16* __restrict__ frag,
                                         int w, int nt, int lane) {
    v8f acc = {};
    const bf16* fb = frag + ((size_t)(w * 8 * 16 + nt) * 32 + lane) * 16;
    #pragma unroll
    for (int kt = 0; kt < 8; kt++) {
        v16bf bfr = *(const v16bf*)(fb + (size_t)kt * 16 * 32 * 16);
        acc = __builtin_amdgcn_wmma_f32_16x16x32_bf16(
                false, a[kt], false, bfr, (short)0, acc, false, false);
    }
    return acc;
}

__global__ void qkv_kernel(const bf16* __restrict__ xnb, const bf16* __restrict__ frag,
                           const float* __restrict__ bq, const float* __restrict__ bk,
                           const float* __restrict__ bv,
                           bf16* __restrict__ qb, bf16* __restrict__ kb, bf16* __restrict__ vtb) {
    int b = blockIdx.x >> 3;
    int tb = blockIdx.x & 7;
    int wave = threadIdx.x >> 5, lane = threadIdx.x & 31;
    int t0 = tb * 128 + wave * 16;
    int lrow = lane & 15, lhalf = lane >> 4;
    const bf16* xrow = xnb + ((size_t)b * NTOK + t0 + lrow) * DIM;
    v16bf a[8];
    #pragma unroll
    for (int kt = 0; kt < 8; kt++) {
        union { v16bf v; v8bf h[2]; } u;
        u.h[0] = *(const v8bf*)(xrow + kt * 32 + lhalf * 16);
        u.h[1] = *(const v8bf*)(xrow + kt * 32 + lhalf * 16 + 8);
        a[kt] = u.v;
    }
    // ---- Q (scale 1/16 folded into packed Wq and bias here) ----
    for (int nt = 0; nt < 16; nt++) {
        v8f acc = gemm_tile(a, frag, 0, nt, lane);
        int cout = nt * 16 + lrow;
        float bval = bq[cout] * 0.0625f;
        int head = cout >> 6, d = cout & 63;
        bf16* dst = qb + ((size_t)(b * NHEAD + head) * NTOK + t0 + lhalf * 8) * DH + d;
        #pragma unroll
        for (int r = 0; r < 8; r++) dst[(size_t)r * DH] = f2bf(acc[r] + bval);
    }
    // ---- K ----
    for (int nt = 0; nt < 16; nt++) {
        v8f acc = gemm_tile(a, frag, 1, nt, lane);
        int cout = nt * 16 + lrow;
        float bval = bk[cout];
        int head = cout >> 6, d = cout & 63;
        bf16* dst = kb + ((size_t)(b * NHEAD + head) * NTOK + t0 + lhalf * 8) * DH + d;
        #pragma unroll
        for (int r = 0; r < 8; r++) dst[(size_t)r * DH] = f2bf(acc[r] + bval);
    }
    // ---- V (stored transposed: [d][tok]) ----
    for (int nt = 0; nt < 16; nt++) {
        v8f acc = gemm_tile(a, frag, 2, nt, lane);
        int cout = nt * 16 + lrow;
        float bval = bv[cout];
        int head = cout >> 6, d = cout & 63;
        bf16* dst = vtb + ((size_t)(b * NHEAD + head) * DH + d) * NTOK + t0 + lhalf * 8;
        #pragma unroll
        for (int r = 0; r < 8; r++) dst[r] = f2bf(acc[r] + bval);
    }
}

// ---------------------------------------------------------------------------
// Kernel 4: fused flash attention per (b,h). Each wave owns 16 query rows;
// streams 32 keys/iter: 4 WMMAs scores -> LDS online softmax -> 4 WMMAs P*V.
// Softmax row state is replicated across lane pairs (L, L+16): each lane of a
// pair reduces 16 of the 32 columns, combined with one shfl_xor -> no EXEC
// divergence and half the serial exp chain.
// ---------------------------------------------------------------------------
__global__ void attn_kernel(const bf16* __restrict__ qb, const bf16* __restrict__ kb,
                            const bf16* __restrict__ vtb, bf16* __restrict__ hb) {
    __shared__ float sS[8][16 * SROW];
    __shared__ bf16  sP[8][16 * PROW];
    int bh = blockIdx.x >> 3;          // b*4 + h
    int qblk = blockIdx.x & 7;
    int wave = threadIdx.x >> 5, lane = threadIdx.x & 31;
    int q0 = qblk * 128 + wave * 16;
    int lrow = lane & 15, lhalf = lane >> 4;
    float* sSw = sS[wave];
    bf16*  sPw = sP[wave];

    const bf16* qrow = qb + ((size_t)bh * NTOK + q0 + lrow) * DH;
    v16bf qa[2];
    #pragma unroll
    for (int kt = 0; kt < 2; kt++) {
        union { v16bf v; v8bf h[2]; } u;
        u.h[0] = *(const v8bf*)(qrow + kt * 32 + lhalf * 16);
        u.h[1] = *(const v8bf*)(qrow + kt * 32 + lhalf * 16 + 8);
        qa[kt] = u.v;
    }
    v8f o0 = {}, o1 = {}, o2 = {}, o3 = {};
    float m = -3.0e38f, l = 0.f;     // row state, replicated in lanes L and L+16
    const bf16* vbase = vtb + (size_t)bh * DH * NTOK;

    for (int jc = 0; jc < 32; jc++) {
        int j0 = jc * 32;
        // ---- scores: S(16x32) = Q(16x64) * K^T ----
        #pragma unroll
        for (int t = 0; t < 2; t++) {
            v8f s = {};
            const bf16* krow = kb + ((size_t)bh * NTOK + j0 + t * 16 + lrow) * DH;
            #pragma unroll
            for (int kt = 0; kt < 2; kt++) {
                v16bf kf = *(const v16bf*)(krow + kt * 32 + lhalf * 16);
                s = __builtin_amdgcn_wmma_f32_16x16x32_bf16(
                        false, qa[kt], false, kf, (short)0, s, false, false);
            }
            #pragma unroll
            for (int r = 0; r < 8; r++)
                sSw[(r + lhalf * 8) * SROW + t * 16 + lrow] = s[r];
        }
        asm volatile("s_wait_dscnt 0" ::: "memory");
        // ---- online softmax: lane pair (lrow, lrow+16) owns row lrow ----
        {
            const float* sr = sSw + lrow * SROW + lhalf * 16;
            float rmax = -3.0e38f;
            #pragma unroll
            for (int i = 0; i < 16; i++) rmax = fmaxf(rmax, sr[i]);
            rmax = fmaxf(rmax, __shfl_xor(rmax, 16, 32));
            float mn = fmaxf(m, rmax);
            float corr = __expf(m - mn);
            float psum = 0.f;
            bf16* pr = sPw + lrow * PROW + lhalf * 16;
            #pragma unroll
            for (int i = 0; i < 16; i++) {
                float e = __expf(sr[i] - mn);
                psum += e;
                pr[i] = f2bf(e);
            }
            psum += __shfl_xor(psum, 16, 32);
            l = l * corr + psum;
            m = mn;
            asm volatile("s_wait_dscnt 0" ::: "memory");
            // ---- rescale accumulators by per-row correction ----
            float cf[8];
            #pragma unroll
            for (int r = 0; r < 8; r++) cf[r] = __shfl(corr, lhalf * 8 + r, 32);
            #pragma unroll
            for (int r = 0; r < 8; r++) { o0[r] *= cf[r]; o1[r] *= cf[r];
                                          o2[r] *= cf[r]; o3[r] *= cf[r]; }
        }
        // ---- P A-fragment from LDS ----
        union { v16bf v; v8bf h[2]; } up;
        up.h[0] = *(const v8bf*)(sPw + lrow * PROW + lhalf * 16);
        up.h[1] = *(const v8bf*)(sPw + lrow * PROW + lhalf * 16 + 8);
        v16bf pa = up.v;
        // ---- O(16x64) += P(16x32) * V(32x64) ----
        v16bf vf0 = *(const v16bf*)(vbase + (size_t)(0 * 16 + lrow) * NTOK + j0 + lhalf * 16);
        o0 = __builtin_amdgcn_wmma_f32_16x16x32_bf16(false, pa, false, vf0, (short)0, o0, false, false);
        v16bf vf1 = *(const v16bf*)(vbase + (size_t)(1 * 16 + lrow) * NTOK + j0 + lhalf * 16);
        o1 = __builtin_amdgcn_wmma_f32_16x16x32_bf16(false, pa, false, vf1, (short)0, o1, false, false);
        v16bf vf2 = *(const v16bf*)(vbase + (size_t)(2 * 16 + lrow) * NTOK + j0 + lhalf * 16);
        o2 = __builtin_amdgcn_wmma_f32_16x16x32_bf16(false, pa, false, vf2, (short)0, o2, false, false);
        v16bf vf3 = *(const v16bf*)(vbase + (size_t)(3 * 16 + lrow) * NTOK + j0 + lhalf * 16);
        o3 = __builtin_amdgcn_wmma_f32_16x16x32_bf16(false, pa, false, vf3, (short)0, o3, false, false);
    }
    // ---- normalize by row sum, write h token-major bf16 ----
    float lf[8];
    #pragma unroll
    for (int r = 0; r < 8; r++) lf[r] = __shfl(l, lhalf * 8 + r, 32);
    int b = bh >> 2, h = bh & 3;
    #pragma unroll
    for (int r = 0; r < 8; r++) {
        float inv = 1.0f / lf[r];
        int tok = q0 + r + lhalf * 8;
        bf16* hp = hb + ((size_t)b * NTOK + tok) * DIM + h * DH;
        hp[0 * 16 + lrow] = f2bf(o0[r] * inv);
        hp[1 * 16 + lrow] = f2bf(o1[r] * inv);
        hp[2 * 16 + lrow] = f2bf(o2[r] * inv);
        hp[3 * 16 + lrow] = f2bf(o3[r] * inv);
    }
}

// ---------------------------------------------------------------------------
// Kernel 5: output projection GEMM + residual: out = (x + h*Wp + bp) / sqrt(2)
// ---------------------------------------------------------------------------
__global__ void proj_kernel(const bf16* __restrict__ hb, const bf16* __restrict__ frag,
                            const float* __restrict__ bp, const float* __restrict__ x,
                            float* __restrict__ out) {
    int b = blockIdx.x >> 3;
    int tb = blockIdx.x & 7;
    int wave = threadIdx.x >> 5, lane = threadIdx.x & 31;
    int t0 = tb * 128 + wave * 16;
    int lrow = lane & 15, lhalf = lane >> 4;
    const bf16* hrow = hb + ((size_t)b * NTOK + t0 + lrow) * DIM;
    v16bf a[8];
    #pragma unroll
    for (int kt = 0; kt < 8; kt++) {
        union { v16bf v; v8bf h[2]; } u;
        u.h[0] = *(const v8bf*)(hrow + kt * 32 + lhalf * 16);
        u.h[1] = *(const v8bf*)(hrow + kt * 32 + lhalf * 16 + 8);
        a[kt] = u.v;
    }
    const float rs2 = 0.70710678118654752f;
    for (int nt = 0; nt < 16; nt++) {
        v8f acc = gemm_tile(a, frag, 3, nt, lane);
        int c = nt * 16 + lrow;
        float bias = bp[c];
        const float* xp = x + ((size_t)b * DIM + c) * NTOK + t0 + lhalf * 8;
        float* op = out + ((size_t)b * DIM + c) * NTOK + t0 + lhalf * 8;
        #pragma unroll
        for (int r = 0; r < 8; r++)
            op[r] = (xp[r] + acc[r] + bias) * rs2;
    }
}

// ---------------------------------------------------------------------------
extern "C" void kernel_launch(void* const* d_in, const int* in_sizes, int n_in,
                              void* d_out, int out_size, void* d_ws, size_t ws_size,
                              hipStream_t stream) {
    const float* x        = (const float*)d_in[0];
    const float* gn_scale = (const float*)d_in[1];
    const float* gn_bias  = (const float*)d_in[2];
    const float* Wq = (const float*)d_in[3];
    const float* bq = (const float*)d_in[4];
    const float* Wk = (const float*)d_in[5];
    const float* bk = (const float*)d_in[6];
    const float* Wv = (const float*)d_in[7];
    const float* bv = (const float*)d_in[8];
    const float* Wp = (const float*)d_in[9];
    const float* bp = (const float*)d_in[10];
    float* out = (float*)d_out;

    char* ws = (char*)d_ws;
    const size_t SZ_ACT  = (size_t)NBATCH * NTOK * DIM * sizeof(bf16);   // 8 MB
    const size_t SZ_FRAG = (size_t)4 * 8 * 16 * 32 * 16 * sizeof(bf16);  // 512 KB
    bf16* xnb  = (bf16*)(ws);                 // also reused as hb (attn output)
    bf16* frag = (bf16*)(ws + SZ_ACT);
    bf16* qb   = (bf16*)(ws + SZ_ACT + SZ_FRAG);
    bf16* kb   = (bf16*)(ws + SZ_ACT + SZ_FRAG + SZ_ACT);
    bf16* vtb  = (bf16*)(ws + SZ_ACT + SZ_FRAG + 2 * SZ_ACT);
    bf16* hb   = xnb;  // xnb dead after qkv_kernel; reuse for attention output

    gn_kernel  <<<NBATCH * 32, 256, 0, stream>>>(x, gn_scale, gn_bias, xnb);
    pack_kernel<<<64,          256, 0, stream>>>(Wq, Wk, Wv, Wp, frag);
    qkv_kernel <<<NBATCH * 8,  256, 0, stream>>>(xnb, frag, bq, bk, bv, qb, kb, vtb);
    attn_kernel<<<NBATCH * NHEAD * 8, 256, 0, stream>>>(qb, kb, vtb, hb);
    proj_kernel<<<NBATCH * 8,  256, 0, stream>>>(hb, frag, bp, x, out);
}
